// Net_66133906424358
// MI455X (gfx1250) — matrix operand order repaired and checked by reading, not compiled, not run on hardware
//
#include <hip/hip_runtime.h>
#include <hip/hip_bf16.h>

// ---------------------------------------------------------------------------
// PointTransformerConv x4 + MLP head for MI455X (gfx1250).
// All dense / gathered GEMMs run on v_wmma_f32_16x16x32_bf16 (f32 accum).
// Segment-softmax passes are bandwidth/atomic bound (L2-resident node tables).
//
// Workspace layout (bytes), all row-major f32 unless noted:
//   Xcat   : [N,288]  padded concat (cols 0-5 x, 6-69 x1, 70-133 x2,
//                      134-197 x3, 198-261 x4, 262-287 zero pad)
//   h      : [N,64]   value projection (reused as MLP1 hidden at the end)
//   aS,aD  : [N,64]   attention src/dst projections
//   amaxk  : [N,64]   u32 order-preserving keys for float segment max
//   denom  : [N,64]   softmax denominators
//   delta  : [E,64]   pos_nn(x_i - x_j)
//   alpha  : [E,64]   logits, overwritten in-place with exp() weights
// Total ~550 MB.
// ---------------------------------------------------------------------------

typedef __bf16 bf16_t;
typedef __attribute__((ext_vector_type(16))) bf16_t v16bf;
typedef __attribute__((ext_vector_type(8)))  float  v8f;

#define LDX   288
#define KPMAX 288

// Order-preserving float -> u32 key (monotonic): any finite float maps to a
// key > 0, so amax keys can be zero-initialized.
__device__ __forceinline__ unsigned fkey(float f) {
  unsigned b = __float_as_uint(f);
  return (b & 0x80000000u) ? ~b : (b | 0x80000000u);
}
__device__ __forceinline__ float fdec(unsigned k) {
  unsigned b = (k & 0x80000000u) ? (k & 0x7FFFFFFFu) : ~k;
  return __uint_as_float(b);
}
__device__ __forceinline__ void pack4(v16bf& a, int o, float4 v) {
  a[o + 0] = (bf16_t)v.x; a[o + 1] = (bf16_t)v.y;
  a[o + 2] = (bf16_t)v.z; a[o + 3] = (bf16_t)v.w;
}
__device__ __forceinline__ float4 f4sub(float4 a, float4 b) {
  return make_float4(a.x - b.x, a.y - b.y, a.z - b.z, a.w - b.w);
}

// ---------------------------------------------------------------------------
// Stage W[K,64] into LDS transposed & zero-padded as bf16: Wt[n*KP + k].
// KP is a multiple of 32, so 64*KP is a multiple of 128 (no divergence).
// ---------------------------------------------------------------------------
__device__ __forceinline__ void stage_w(bf16_t* Wt, const float* __restrict__ W,
                                        int K, int KP) {
  for (int idx = threadIdx.x; idx < 64 * KP; idx += 128) {
    int n = idx / KP, k = idx - n * KP;
    Wt[idx] = (bf16_t)((k < K) ? W[(size_t)k * 64 + n] : 0.0f);
  }
  __syncthreads();
}

// Load this lane's B fragment for column tile t (col = 16t + lane%16).
// B k order used here (contiguous, same as A): lanes 0-15 hold K=k0..k0+15, lanes 16-31 K=k0+16..k0+31,
// contiguous within the lane -> one 32B LDS read.
__device__ __forceinline__ v16bf load_b(const bf16_t* Wt, int KP, int k0,
                                        int lm, int half, int t) {
  return *(const v16bf*)&Wt[(size_t)(t * 16 + lm) * KP + k0 + half * 16];
}

// Build this lane's A fragment from a row pointer (contiguous k order, same as B:
// half 0 -> K = k0+{0..15}; half 1 -> K = k0+{16..31}).
__device__ __forceinline__ v16bf load_a_row(const float* __restrict__ ap,
                                            int k0, int half) {
  const float4* p = (const float4*)(ap + k0 + half * 16);
  float4 x0 = p[0], x1 = p[1], x2 = p[2], x3 = p[3];
  v16bf a;
  pack4(a, 0, x0); pack4(a, 4, x1); pack4(a, 8, x2); pack4(a, 12, x3);
  return a;
}

// ---------------------------------------------------------------------------
// Y[row, col0 + 0..63] = A[row, 0..K) @ W[K,64] (+ bias). 128 threads = 4
// waves; each wave owns a 16x64 tile (4 accumulators). A is f32 with row
// stride lda, padded so cols [K, KP) are finite (zeros) -> no K guards.
// ---------------------------------------------------------------------------
__global__ void __launch_bounds__(128)
gemm_bf16_wmma(const float* __restrict__ A, int lda, int M,
               const float* __restrict__ W, int K, int KP,
               const float* __restrict__ bias,
               float* __restrict__ Y, int ldy, int col0) {
  __shared__ __align__(64) bf16_t Wt[64 * KPMAX];
  stage_w(Wt, W, K, KP);

  int lane = threadIdx.x & 31, wv = threadIdx.x >> 5;
  int half = lane >> 4, lm = lane & 15;

  int row = blockIdx.x * 64 + wv * 16 + lm;
  if (row >= M) row = M - 1;                       // clamp; stores guarded
  const float* ap = A + (size_t)row * lda;

  v8f acc[4] = {};
  for (int k0 = 0; k0 < KP; k0 += 32) {
    v16bf a = load_a_row(ap, k0, half);
#pragma unroll
    for (int t = 0; t < 4; ++t) {
      v16bf b = load_b(Wt, KP, k0, lm, half, t);
      acc[t] = __builtin_amdgcn_wmma_f32_16x16x32_bf16(
          false, a, false, b, (short)0, acc[t], false, false);
    }
  }

  int mrow0 = blockIdx.x * 64 + wv * 16 + half * 8;   // D rows for this lane
#pragma unroll
  for (int t = 0; t < 4; ++t) {
    float bv = bias ? bias[t * 16 + lm] : 0.0f;
#pragma unroll
    for (int r = 0; r < 8; ++r) {
      int rr = mrow0 + r;
      if (rr < M)
        Y[(size_t)rr * ldy + col0 + t * 16 + lm] = acc[t][r] + bv;
    }
  }
}

// ---------------------------------------------------------------------------
// Per-edge: delta = (x[dst]-x[src]) @ w_pos + b_pos (WMMA on gathered rows),
// fused with alpha = delta + a_dst[dst] - a_src[src], alpha store, and the
// ordered-key atomicMax for the segment max. Each wave owns 16 edges x 64 ch.
// ---------------------------------------------------------------------------
__global__ void __launch_bounds__(128)
edge_delta_alpha_wmma(const float* __restrict__ Xcat,
                      const float* __restrict__ Wpos, int K, int KP,
                      const float* __restrict__ bpos,
                      const int* __restrict__ srcIdx,
                      const int* __restrict__ dstIdx,
                      const float* __restrict__ aS,
                      const float* __restrict__ aD,
                      float* __restrict__ delta, float* __restrict__ alpha,
                      unsigned* __restrict__ amaxk, int E) {
  __shared__ __align__(64) bf16_t Wt[64 * KPMAX];
  stage_w(Wt, Wpos, K, KP);

  int lane = threadIdx.x & 31, wv = threadIdx.x >> 5;
  int half = lane >> 4, lm = lane & 15;
  int eb = blockIdx.x * 64 + wv * 16;

  int e = eb + lm; if (e >= E) e = E - 1;          // clamp; stores guarded
  const float* xs = Xcat + (size_t)srcIdx[e] * LDX;
  const float* xd = Xcat + (size_t)dstIdx[e] * LDX;

  v8f acc[4] = {};
  for (int k0 = 0; k0 < KP; k0 += 32) {
    const float4* pd = (const float4*)(xd + k0 + half * 16);
    const float4* ps = (const float4*)(xs + k0 + half * 16);
    v16bf a;
    pack4(a, 0,  f4sub(pd[0], ps[0]));
    pack4(a, 4,  f4sub(pd[1], ps[1]));
    pack4(a, 8,  f4sub(pd[2], ps[2]));
    pack4(a, 12, f4sub(pd[3], ps[3]));
#pragma unroll
    for (int t = 0; t < 4; ++t) {
      v16bf b = load_b(Wt, KP, k0, lm, half, t);
      acc[t] = __builtin_amdgcn_wmma_f32_16x16x32_bf16(
          false, a, false, b, (short)0, acc[t], false, false);
    }
  }

  // Edge ids for this lane's 8 output rows (D layout: m = 8*half + r).
  int rows0 = eb + half * 8;
  int rid_s[8], rid_d[8];
#pragma unroll
  for (int r = 0; r < 8; ++r) {
    int er = rows0 + r; if (er >= E) er = E - 1;
    rid_s[r] = srcIdx[er]; rid_d[r] = dstIdx[er];
  }
#pragma unroll
  for (int t = 0; t < 4; ++t) {
    int n = t * 16 + lm;
    float bv = bpos[n];
#pragma unroll
    for (int r = 0; r < 8; ++r) {
      int er = rows0 + r;
      if (er < E) {
        float dval = acc[t][r] + bv;
        delta[(size_t)er * 64 + n] = dval;
        float aval = dval + aD[(size_t)rid_d[r] * 64 + n]
                          - aS[(size_t)rid_s[r] * 64 + n];
        alpha[(size_t)er * 64 + n] = aval;
        atomicMax(&amaxk[(size_t)rid_d[r] * 64 + n], fkey(aval));
      }
    }
  }
}

// ---------------------------------------------------------------------------
// Elementwise edge passes.
// ---------------------------------------------------------------------------
__global__ void ek_exp(const int* __restrict__ dstIdx,
                       const unsigned* __restrict__ amaxk,
                       float* __restrict__ alpha_ew,
                       float* __restrict__ denom, long long total) {
  long long idx = (long long)blockIdx.x * blockDim.x + threadIdx.x;
  if (idx >= total) return;
  int e = (int)(idx >> 6), n = (int)(idx & 63);
  int d = dstIdx[e];
  float ev = __expf(alpha_ew[idx] - fdec(amaxk[(size_t)d * 64 + n]));
  alpha_ew[idx] = ev;
  atomicAdd(&denom[(size_t)d * 64 + n], ev);
}

__global__ void ek_agg(const int* __restrict__ srcIdx,
                       const int* __restrict__ dstIdx,
                       const float* __restrict__ ew,
                       const float* __restrict__ delta,
                       const float* __restrict__ h,
                       const float* __restrict__ denom,
                       float* __restrict__ Xcat, int col0, long long total) {
  long long idx = (long long)blockIdx.x * blockDim.x + threadIdx.x;
  if (idx >= total) return;
  int e = (int)(idx >> 6), n = (int)(idx & 63);
  int d = dstIdx[e], s = srcIdx[e];
  float w = ew[idx] / (denom[(size_t)d * 64 + n] + 1e-16f);
  float v = w * (h[(size_t)s * 64 + n] + delta[idx]);
  atomicAdd(&Xcat[(size_t)d * LDX + col0 + n], v);
}

// ---------------------------------------------------------------------------
// Small utility kernels.
// ---------------------------------------------------------------------------
__global__ void init_xcat(const float* __restrict__ x, float* __restrict__ Xcat,
                          long long total) {
  long long idx = (long long)blockIdx.x * blockDim.x + threadIdx.x;
  if (idx >= total) return;
  int i = (int)(idx / LDX), c = (int)(idx - (long long)i * LDX);
  Xcat[idx] = (c < 6) ? x[(size_t)i * 6 + c] : 0.0f;
}

__global__ void zero_nd(unsigned* __restrict__ amaxk, float* __restrict__ denom,
                        long long total) {
  long long idx = (long long)blockIdx.x * blockDim.x + threadIdx.x;
  if (idx >= total) return;
  amaxk[idx] = 0u;
  denom[idx] = 0.0f;
}

__global__ void lrelu_block(float* __restrict__ Xcat, int col0, long long total) {
  long long idx = (long long)blockIdx.x * blockDim.x + threadIdx.x;
  if (idx >= total) return;
  int i = (int)(idx >> 6), n = (int)(idx & 63);
  float* p = &Xcat[(size_t)i * LDX + col0 + n];
  float v = *p;
  *p = v > 0.0f ? v : 0.01f * v;
}

__global__ void mlp2_kernel(const float* __restrict__ hid,
                            const float* __restrict__ w2,
                            const float* __restrict__ b2,
                            float* __restrict__ out, long long total) {
  long long idx = (long long)blockIdx.x * blockDim.x + threadIdx.x;
  if (idx >= total) return;
  int i = (int)(idx / 10), j = (int)(idx - (long long)i * 10);
  float s = b2[j];
  const float* hp = hid + (size_t)i * 64;
#pragma unroll
  for (int k = 0; k < 64; ++k) s += hp[k] * w2[k * 10 + j];
  out[idx] = s;
}

// ---------------------------------------------------------------------------
// Host orchestration.
// ---------------------------------------------------------------------------
extern "C" void kernel_launch(void* const* d_in, const int* in_sizes, int n_in,
                              void* d_out, int out_size, void* d_ws,
                              size_t ws_size, hipStream_t stream) {
  const float* x   = (const float*)d_in[0];
  const int* eidx  = (const int*)d_in[1];
  const int N = in_sizes[0] / 6;
  const int E = in_sizes[1] / 2;
  const int* srcIdx = eidx;       // edge_index[0] = src
  const int* dstIdx = eidx + E;   // edge_index[1] = dst

  const float *w_lin[4], *w_src[4], *w_dst[4], *w_pos[4], *b_pos[4];
  int p = 2;
  for (int l = 0; l < 4; ++l) {
    w_lin[l] = (const float*)d_in[p++];
    w_src[l] = (const float*)d_in[p++];
    w_dst[l] = (const float*)d_in[p++];
    w_pos[l] = (const float*)d_in[p++];
    b_pos[l] = (const float*)d_in[p++];
  }
  const float* w_mlp1 = (const float*)d_in[p++];
  const float* b_mlp1 = (const float*)d_in[p++];
  const float* w_mlp2 = (const float*)d_in[p++];
  const float* b_mlp2 = (const float*)d_in[p++];

  // Workspace carving (~550 MB for N=100k, E=600k).
  char* ws = (char*)d_ws;
  float* Xcat = (float*)ws;      ws += (size_t)N * LDX * sizeof(float);
  float* h    = (float*)ws;      ws += (size_t)N * 64 * sizeof(float);
  float* aS   = (float*)ws;      ws += (size_t)N * 64 * sizeof(float);
  float* aD   = (float*)ws;      ws += (size_t)N * 64 * sizeof(float);
  unsigned* amaxk = (unsigned*)ws; ws += (size_t)N * 64 * sizeof(unsigned);
  float* denom = (float*)ws;     ws += (size_t)N * 64 * sizeof(float);
  float* delta = (float*)ws;     ws += (size_t)E * 64 * sizeof(float);
  float* alpha = (float*)ws;     // reused in place as exp-weights

  const long long nXc = (long long)N * LDX;
  const long long nN64 = (long long)N * 64;
  const long long nE64 = (long long)E * 64;
  const int T = 256;
  const int gemmBlocks = (N + 63) / 64;
  const int edgeBlocks = (E + 63) / 64;

  init_xcat<<<(unsigned)((nXc + T - 1) / T), T, 0, stream>>>(x, Xcat, nXc);

  const int FIN[4] = {6, 70, 134, 198};
  for (int l = 0; l < 4; ++l) {
    const int K = FIN[l], KP = ((K + 31) / 32) * 32;
    const int colOut = 6 + 64 * l;

    gemm_bf16_wmma<<<gemmBlocks, 128, 0, stream>>>(
        Xcat, LDX, N, w_lin[l], K, KP, nullptr, h, 64, 0);
    gemm_bf16_wmma<<<gemmBlocks, 128, 0, stream>>>(
        Xcat, LDX, N, w_src[l], K, KP, nullptr, aS, 64, 0);
    gemm_bf16_wmma<<<gemmBlocks, 128, 0, stream>>>(
        Xcat, LDX, N, w_dst[l], K, KP, nullptr, aD, 64, 0);

    zero_nd<<<(unsigned)((nN64 + T - 1) / T), T, 0, stream>>>(amaxk, denom, nN64);

    edge_delta_alpha_wmma<<<edgeBlocks, 128, 0, stream>>>(
        Xcat, w_pos[l], K, KP, b_pos[l], srcIdx, dstIdx, aS, aD,
        delta, alpha, amaxk, E);

    ek_exp<<<(unsigned)((nE64 + T - 1) / T), T, 0, stream>>>(
        dstIdx, amaxk, alpha, denom, nE64);

    ek_agg<<<(unsigned)((nE64 + T - 1) / T), T, 0, stream>>>(
        srcIdx, dstIdx, alpha, delta, h, denom, Xcat, colOut, nE64);

    if (l >= 1)
      lrelu_block<<<(unsigned)((nN64 + T - 1) / T), T, 0, stream>>>(
          Xcat, colOut, nN64);
  }

  // MLP head: [N,262] @ [262,64] + b (WMMA, K padded to 288), then 64->10.
  gemm_bf16_wmma<<<gemmBlocks, 128, 0, stream>>>(
      Xcat, LDX, N, w_mlp1, 262, 288, b_mlp1, h, 64, 0);
  const long long nOut = (long long)N * 10;
  mlp2_kernel<<<(unsigned)((nOut + T - 1) / T), T, 0, stream>>>(
      h, w_mlp2, b_mlp2, (float*)d_out, nOut);
}
